// MultiHeadAttention_50311246905859
// MI455X (gfx1250) — matrix operand order repaired and checked by reading, not compiled, not run on hardware
//
#include <hip/hip_runtime.h>

// ---------------------------------------------------------------------------
// MultiHeadAttention for MI455X (gfx1250, wave32, WMMA).
// Pipeline: f32->bf16 convert | fused QKV GEMM (WMMA bf16) |
//           flash-attention (WMMA bf16, online softmax) | output GEMM (WMMA).
// GEMMs use 32x64 register tiles per wave (8 WMMAs / 12 b128 loads per k-step).
// ---------------------------------------------------------------------------

typedef __attribute__((ext_vector_type(16))) __bf16 v16bf;
typedef __attribute__((ext_vector_type(8)))  float  v8f;

union BF16x16 { v16bf v; uint4 q[2]; };

__device__ __forceinline__ v8f wmma_bf16(v16bf a, v16bf b, v8f c) {
    return __builtin_amdgcn_wmma_f32_16x16x32_bf16(
        /*neg_a=*/false, a, /*neg_b=*/false, b,
        /*c_mod=*/(short)0, c, /*reuse_a=*/false, /*reuse_b=*/false);
}

#define BATCH    4
#define SEQ      2048
#define DM       1024
#define NH       16
#define HD       64
#define MTOT     (BATCH * SEQ)     // 8192 rows of x

// ---------------------------------------------------------------------------
// fp32 -> bf16 elementwise convert (vectorized x4)
// ---------------------------------------------------------------------------
__global__ __launch_bounds__(256) void cvt_f32_bf16(
    const float* __restrict__ src, __bf16* __restrict__ dst, int n4) {
    int i = blockIdx.x * blockDim.x + threadIdx.x;
    if (i < n4) {
        float4 f = reinterpret_cast<const float4*>(src)[i];
        __bf16* o = dst + (size_t)i * 4;
        o[0] = (__bf16)f.x; o[1] = (__bf16)f.y;
        o[2] = (__bf16)f.z; o[3] = (__bf16)f.w;
    }
}

// ---------------------------------------------------------------------------
// QKV projection: C[m,n] = sum_k X[m,k]*W[n,k] + bias[n]
// Each wave computes a 32x64 tile (2x4 WMMA tiles); 4 waves -> 64x128 / block.
// mode 0/1: write (B,H,T,64) head-major (Q,K). mode 2: write (B,H,64,T) (V^T).
// ---------------------------------------------------------------------------
__global__ __launch_bounds__(128) void qkv_gemm(
    const __bf16* __restrict__ X, const __bf16* __restrict__ W,
    const float* __restrict__ bias, __bf16* __restrict__ dst, int mode) {

    const int lane = threadIdx.x & 31;
    const int wave = threadIdx.x >> 5;
    const int l15  = lane & 15;
    const bool lt16 = lane < 16;

    const int Mbase = blockIdx.y * 64 + (wave >> 1) * 32;
    const int Nbase = blockIdx.x * 128 + (wave & 1) * 64;

    const __bf16* arow0 = X + (size_t)(Mbase + l15) * DM;
    const __bf16* arow1 = arow0 + (size_t)16 * DM;
    const __bf16* brow[4];
    #pragma unroll
    for (int nt = 0; nt < 4; ++nt)
        brow[nt] = W + (size_t)(Nbase + nt * 16 + l15) * DM;

    const int aoff = lt16 ? 0 : 16;  // A frag: K {0..15} / {16..31}
    const int boff = lt16 ? 0 : 16;  // B frag: K {0..15} / {16..31}

    v8f acc[2][4];
    #pragma unroll
    for (int mi = 0; mi < 2; ++mi)
        #pragma unroll
        for (int ni = 0; ni < 4; ++ni) acc[mi][ni] = (v8f){};

    for (int k0 = 0; k0 < DM; k0 += 32) {
        BF16x16 a0, a1, b[4];
        a0.q[0] = *reinterpret_cast<const uint4*>(arow0 + k0 + aoff);
        a0.q[1] = *reinterpret_cast<const uint4*>(arow0 + k0 + aoff + 8);
        a1.q[0] = *reinterpret_cast<const uint4*>(arow1 + k0 + aoff);
        a1.q[1] = *reinterpret_cast<const uint4*>(arow1 + k0 + aoff + 8);
        #pragma unroll
        for (int nt = 0; nt < 4; ++nt) {
            b[nt].q[0] = *reinterpret_cast<const uint4*>(brow[nt] + k0 + boff);
            b[nt].q[1] = *reinterpret_cast<const uint4*>(brow[nt] + k0 + boff + 8);
        }
        #pragma unroll
        for (int nt = 0; nt < 4; ++nt) {
            acc[0][nt] = wmma_bf16(a0.v, b[nt].v, acc[0][nt]);
            acc[1][nt] = wmma_bf16(a1.v, b[nt].v, acc[1][nt]);
        }
    }

    // epilogue: scatter to head-major layouts
    #pragma unroll
    for (int mi = 0; mi < 2; ++mi) {
        #pragma unroll
        for (int ni = 0; ni < 4; ++ni) {
            const v8f c = acc[mi][ni];
            const int col = Nbase + ni * 16 + l15;
            const float bv = bias[col];
            const int h = col >> 6, d = col & 63;
            #pragma unroll
            for (int r = 0; r < 8; ++r) {
                const int row = Mbase + mi * 16 + (lt16 ? r : r + 8);
                const int bb = row >> 11;      // row / SEQ
                const int t  = row & (SEQ - 1);
                const float v = c[r] + bv;
                size_t idx;
                if (mode == 2)  // V transposed: (B,H,64,T)
                    idx = ((size_t)(bb * NH + h) * HD + d) * SEQ + t;
                else            // Q,K: (B,H,T,64)
                    idx = ((size_t)(bb * NH + h) * SEQ + t) * HD + d;
                dst[idx] = (__bf16)v;
            }
        }
    }
}

// ---------------------------------------------------------------------------
// Flash attention: one wave handles a 16-query tile for one (b,h).
// Keys processed 32 at a time: 4 score WMMAs, causal mask, online softmax,
// LDS transpose of the 16x32 P tile into A-fragment layout, 4 PV WMMAs.
// ---------------------------------------------------------------------------
__global__ __launch_bounds__(32) void attn_fwd(
    const __bf16* __restrict__ Q, const __bf16* __restrict__ K,
    const __bf16* __restrict__ Vt, __bf16* __restrict__ O) {

    const int lane = threadIdx.x;
    const int l15  = lane & 15;
    const bool lt16 = lane < 16;
    const int bh = blockIdx.y;              // b*NH + h
    const int q0 = blockIdx.x * 16;

    const __bf16* Qb = Q  + (size_t)bh * SEQ * HD;
    const __bf16* Kb = K  + (size_t)bh * SEQ * HD;
    const __bf16* Vb = Vt + (size_t)bh * HD * SEQ;

    const int aoff = lt16 ? 0 : 16;
    const __bf16* qrow = Qb + (size_t)(q0 + l15) * HD;
    BF16x16 aq0, aq1;                        // Q fragments: d 0..31, 32..63
    aq0.q[0] = *reinterpret_cast<const uint4*>(qrow + aoff);
    aq0.q[1] = *reinterpret_cast<const uint4*>(qrow + aoff + 8);
    aq1.q[0] = *reinterpret_cast<const uint4*>(qrow + 32 + aoff);
    aq1.q[1] = *reinterpret_cast<const uint4*>(qrow + 32 + aoff + 8);

    v8f o0 = {}, o1 = {}, o2 = {}, o3 = {};
    float mrow[8], lrow[8];
    #pragma unroll
    for (int r = 0; r < 8; ++r) { mrow[r] = -1e30f; lrow[r] = 0.0f; }

    __shared__ __align__(16) __bf16 Pld[16 * 32];

    const float scale = 0.125f;              // 1/sqrt(64)
    const int qhi = q0 + 15;
    const int koff = lt16 ? 0 : 16;

    for (int j0 = 0; j0 <= qhi; j0 += 32) {
        // ---- scores: S = Q @ K^T for 32 keys (two 16-wide N tiles) ----
        v8f c0 = {}, c1 = {};
        const __bf16* krow0 = Kb + (size_t)(j0 + l15) * HD + koff;
        const __bf16* krow1 = krow0 + (size_t)16 * HD;
        BF16x16 kb;
        kb.q[0] = *reinterpret_cast<const uint4*>(krow0);
        kb.q[1] = *reinterpret_cast<const uint4*>(krow0 + 8);
        c0 = wmma_bf16(aq0.v, kb.v, c0);
        kb.q[0] = *reinterpret_cast<const uint4*>(krow0 + 32);
        kb.q[1] = *reinterpret_cast<const uint4*>(krow0 + 40);
        c0 = wmma_bf16(aq1.v, kb.v, c0);
        kb.q[0] = *reinterpret_cast<const uint4*>(krow1);
        kb.q[1] = *reinterpret_cast<const uint4*>(krow1 + 8);
        c1 = wmma_bf16(aq0.v, kb.v, c1);
        kb.q[0] = *reinterpret_cast<const uint4*>(krow1 + 32);
        kb.q[1] = *reinterpret_cast<const uint4*>(krow1 + 40);
        c1 = wmma_bf16(aq1.v, kb.v, c1);

        // ---- causal mask + online softmax (rows striped across half-wave) --
        const bool partial = (j0 + 31) > q0;
        float p0[8], p1[8];
        #pragma unroll
        for (int r = 0; r < 8; ++r) {
            const int row = q0 + (lt16 ? r : r + 8);
            float s0 = c0[r] * scale;
            float s1 = c1[r] * scale;
            if (partial) {
                if (j0 + l15 > row)      s0 = -1e30f;
                if (j0 + 16 + l15 > row) s1 = -1e30f;
            }
            float mx = fmaxf(s0, s1);
            #pragma unroll
            for (int off = 1; off < 16; off <<= 1)
                mx = fmaxf(mx, __shfl_xor(mx, off, 32));
            const float mnew = fmaxf(mrow[r], mx);
            const float corr = __expf(mrow[r] - mnew);
            mrow[r] = mnew;
            const float e0 = __expf(s0 - mnew);
            const float e1 = __expf(s1 - mnew);
            float rs = e0 + e1;
            #pragma unroll
            for (int off = 1; off < 16; off <<= 1)
                rs += __shfl_xor(rs, off, 32);
            lrow[r] = lrow[r] * corr + rs;
            o0[r] *= corr; o1[r] *= corr; o2[r] *= corr; o3[r] *= corr;
            p0[r] = e0; p1[r] = e1;
        }

        // ---- transpose P (C layout -> A-fragment layout) through LDS ----
        #pragma unroll
        for (int r = 0; r < 8; ++r) {
            const int row = lt16 ? r : r + 8;
            Pld[row * 32 + l15]      = (__bf16)p0[r];
            Pld[row * 32 + 16 + l15] = (__bf16)p1[r];
        }
        __syncthreads();
        BF16x16 ap;
        ap.q[0] = *reinterpret_cast<const uint4*>(Pld + l15 * 32 + aoff);
        ap.q[1] = *reinterpret_cast<const uint4*>(Pld + l15 * 32 + aoff + 8);
        __syncthreads();

        // ---- O += P @ V  (V^T layout -> contiguous B-fragment loads) ----
        BF16x16 vb;
        #pragma unroll
        for (int t = 0; t < 4; ++t) {
            const __bf16* vrow = Vb + (size_t)(t * 16 + l15) * SEQ + j0 + koff;
            vb.q[0] = *reinterpret_cast<const uint4*>(vrow);
            vb.q[1] = *reinterpret_cast<const uint4*>(vrow + 8);
            v8f a = t == 0 ? o0 : t == 1 ? o1 : t == 2 ? o2 : o3;
            a = wmma_bf16(ap.v, vb.v, a);
            if (t == 0) o0 = a; else if (t == 1) o1 = a;
            else if (t == 2) o2 = a; else o3 = a;
        }
    }

    // ---- normalize and write O back to (B,T,DM) row-major (bf16) ----
    const int h = bh & (NH - 1);
    const int bb = bh >> 4;
    #pragma unroll
    for (int r = 0; r < 8; ++r) {
        const float inv = 1.0f / lrow[r];
        const int row = q0 + (lt16 ? r : r + 8);
        const size_t base = ((size_t)bb * SEQ + row) * DM + h * HD + l15;
        O[base + 0]  = (__bf16)(o0[r] * inv);
        O[base + 16] = (__bf16)(o1[r] * inv);
        O[base + 32] = (__bf16)(o2[r] * inv);
        O[base + 48] = (__bf16)(o3[r] * inv);
    }
}

// ---------------------------------------------------------------------------
// Output projection: out[m,n] = sum_k O[m,k]*Wo[n,k] + bo[n]  (fp32 out)
// Same 32x64-per-wave register tiling as qkv_gemm.
// ---------------------------------------------------------------------------
__global__ __launch_bounds__(128) void out_gemm(
    const __bf16* __restrict__ X, const __bf16* __restrict__ W,
    const float* __restrict__ bias, float* __restrict__ out) {

    const int lane = threadIdx.x & 31;
    const int wave = threadIdx.x >> 5;
    const int l15  = lane & 15;
    const bool lt16 = lane < 16;

    const int Mbase = blockIdx.y * 64 + (wave >> 1) * 32;
    const int Nbase = blockIdx.x * 128 + (wave & 1) * 64;

    const __bf16* arow0 = X + (size_t)(Mbase + l15) * DM;
    const __bf16* arow1 = arow0 + (size_t)16 * DM;
    const __bf16* brow[4];
    #pragma unroll
    for (int nt = 0; nt < 4; ++nt)
        brow[nt] = W + (size_t)(Nbase + nt * 16 + l15) * DM;

    const int aoff = lt16 ? 0 : 16;
    const int boff = lt16 ? 0 : 16;

    v8f acc[2][4];
    #pragma unroll
    for (int mi = 0; mi < 2; ++mi)
        #pragma unroll
        for (int ni = 0; ni < 4; ++ni) acc[mi][ni] = (v8f){};

    for (int k0 = 0; k0 < DM; k0 += 32) {
        BF16x16 a0, a1, b[4];
        a0.q[0] = *reinterpret_cast<const uint4*>(arow0 + k0 + aoff);
        a0.q[1] = *reinterpret_cast<const uint4*>(arow0 + k0 + aoff + 8);
        a1.q[0] = *reinterpret_cast<const uint4*>(arow1 + k0 + aoff);
        a1.q[1] = *reinterpret_cast<const uint4*>(arow1 + k0 + aoff + 8);
        #pragma unroll
        for (int nt = 0; nt < 4; ++nt) {
            b[nt].q[0] = *reinterpret_cast<const uint4*>(brow[nt] + k0 + boff);
            b[nt].q[1] = *reinterpret_cast<const uint4*>(brow[nt] + k0 + boff + 8);
        }
        #pragma unroll
        for (int nt = 0; nt < 4; ++nt) {
            acc[0][nt] = wmma_bf16(a0.v, b[nt].v, acc[0][nt]);
            acc[1][nt] = wmma_bf16(a1.v, b[nt].v, acc[1][nt]);
        }
    }

    #pragma unroll
    for (int mi = 0; mi < 2; ++mi) {
        #pragma unroll
        for (int ni = 0; ni < 4; ++ni) {
            const v8f c = acc[mi][ni];
            const int col = Nbase + ni * 16 + l15;
            const float bv = bias[col];
            #pragma unroll
            for (int r = 0; r < 8; ++r) {
                const int row = Mbase + mi * 16 + (lt16 ? r : r + 8);
                out[(size_t)row * DM + col] = c[r] + bv;
            }
        }
    }
}

// ---------------------------------------------------------------------------
// Host launcher. Workspace layout (bytes):
//   xb:  16 MB | Wq,Wk,Wv,Wo bf16: 4x2 MB | Qh,Kh,VtT,Obuf: 4x16 MB  (~88 MB)
// ---------------------------------------------------------------------------
extern "C" void kernel_launch(void* const* d_in, const int* in_sizes, int n_in,
                              void* d_out, int out_size, void* d_ws, size_t ws_size,
                              hipStream_t stream) {
    const float* x  = (const float*)d_in[0];
    const float* Wq = (const float*)d_in[1];
    const float* bq = (const float*)d_in[2];
    const float* Wk = (const float*)d_in[3];
    const float* bk = (const float*)d_in[4];
    const float* Wv = (const float*)d_in[5];
    const float* bv = (const float*)d_in[6];
    const float* Wo = (const float*)d_in[7];
    const float* bo = (const float*)d_in[8];
    float* out = (float*)d_out;

    char* ws = (char*)d_ws;
    const size_t XB = (size_t)MTOT * DM * 2;        // 16 MB
    const size_t WB = (size_t)DM * DM * 2;          //  2 MB
    const size_t HB = (size_t)BATCH * NH * SEQ * HD * 2; // 16 MB
    __bf16* xb  = (__bf16*)(ws);
    __bf16* wqb = (__bf16*)(ws + XB);
    __bf16* wkb = (__bf16*)(ws + XB + WB);
    __bf16* wvb = (__bf16*)(ws + XB + 2 * WB);
    __bf16* wob = (__bf16*)(ws + XB + 3 * WB);
    __bf16* Qh  = (__bf16*)(ws + XB + 4 * WB);
    __bf16* Kh  = (__bf16*)(ws + XB + 4 * WB + HB);
    __bf16* VtT = (__bf16*)(ws + XB + 4 * WB + 2 * HB);
    __bf16* Ob  = (__bf16*)(ws + XB + 4 * WB + 3 * HB);

    // 1) fp32 -> bf16 conversions
    {
        int n4 = MTOT * DM / 4;                     // 2,097,152
        cvt_f32_bf16<<<n4 / 256, 256, 0, stream>>>(x, xb, n4);
        int w4 = DM * DM / 4;                       //   262,144
        cvt_f32_bf16<<<w4 / 256, 256, 0, stream>>>(Wq, wqb, w4);
        cvt_f32_bf16<<<w4 / 256, 256, 0, stream>>>(Wk, wkb, w4);
        cvt_f32_bf16<<<w4 / 256, 256, 0, stream>>>(Wv, wvb, w4);
        cvt_f32_bf16<<<w4 / 256, 256, 0, stream>>>(Wo, wob, w4);
    }

    // 2) QKV projections (64x128 tile per 128-thread block)
    dim3 ggrid(DM / 128, MTOT / 64);                // (8, 128)
    qkv_gemm<<<ggrid, 128, 0, stream>>>(xb, wqb, bq, Qh, 0);
    qkv_gemm<<<ggrid, 128, 0, stream>>>(xb, wkb, bk, Kh, 1);
    qkv_gemm<<<ggrid, 128, 0, stream>>>(xb, wvb, bv, VtT, 2);

    // 3) flash attention: one wave per 16-query tile per (b,h)
    dim3 agrid(SEQ / 16, BATCH * NH);               // (128, 64)
    attn_fwd<<<agrid, 32, 0, stream>>>(Qh, Kh, VtT, Ob);

    // 4) output projection -> fp32 out
    out_gemm<<<ggrid, 128, 0, stream>>>(Ob, wob, bo, out);
}
